// TransformerBlock_72662256714047
// MI455X (gfx1250) — compile-verified, matrix operand order checked by reading, not run on hardware
//
#include <hip/hip_runtime.h>
#include <hip/hip_bf16.h>

#define DEV __device__ __forceinline__

typedef __attribute__((ext_vector_type(16))) __bf16 bf16x16;
typedef __attribute__((ext_vector_type(8)))  float  f32x8;

constexpr int D     = 384;
constexpr int NH    = 6;
constexpr int DH    = 64;
constexpr int HID   = 1536;
constexpr int D3    = 1152;
constexpr int BATCH = 16;
constexpr int SEQ   = 1024;
constexpr int MROWS = BATCH * SEQ; // 16384

// ---------------- helpers ----------------

DEV __bf16 to_bf16(float f) {
  union { float f; unsigned u; } x; x.f = f;
  unsigned r = x.u + 0x7FFFu + ((x.u >> 16) & 1u);
  unsigned short h = (unsigned short)(r >> 16);
  return __builtin_bit_cast(__bf16, h);
}

union FragU { bf16x16 v; uint4 u[2]; };

// Load a 16x32 bf16 A/B-layout fragment from a row-major matrix.
// Per the CDNA5 ISA layout, lane L holds row (L&15); lanes 0-15 carry
// K = {k0..k0+7, k0+16..k0+23}, lanes 16-31 carry K = {k0+8..k0+15, k0+24..k0+31}.
// -> exactly two contiguous 16-byte loads per lane.
DEV bf16x16 load_frag(const __bf16* __restrict__ base, int ld, int row, int k0) {
  const int lane = threadIdx.x & 31;
  const int kb = k0 + ((lane & 16) ? 8 : 0);
  const __bf16* p = base + (size_t)row * (size_t)ld + kb;
  FragU f;
  f.u[0] = *(const uint4*)(p);
  f.u[1] = *(const uint4*)(p + 16);
  return f.v;
}

DEV f32x8 wmma_bf16(bf16x16 a, bf16x16 b, f32x8 c) {
  return __builtin_amdgcn_wmma_f32_16x16x32_bf16(false, a, false, b, (short)0, c,
                                                 false, false);
}

// ---------------- weight conversion ----------------

__global__ void f32_to_bf16_kernel(const float* __restrict__ in,
                                   __bf16* __restrict__ out, int n) {
  int i = blockIdx.x * 256 + threadIdx.x;
  if (i < n) out[i] = to_bf16(in[i]);
}

// ---------------- layernorm (one wave per 384-wide row) ----------------

__global__ __launch_bounds__(256) void ln_rows_kernel(
    const float* __restrict__ in, const float* __restrict__ g,
    const float* __restrict__ bta, __bf16* __restrict__ out) {
  const int lane = threadIdx.x & 31;
  const int wave = threadIdx.x >> 5;
  const int row = blockIdx.x * 8 + wave;
  const float* xr = in + (size_t)row * D;
  float v[12];
  float s = 0.f;
#pragma unroll
  for (int i = 0; i < 12; ++i) { v[i] = xr[lane + i * 32]; s += v[i]; }
#pragma unroll
  for (int m = 16; m >= 1; m >>= 1) s += __shfl_xor(s, m, 32);
  const float mu = s * (1.0f / D);
  float vs = 0.f;
#pragma unroll
  for (int i = 0; i < 12; ++i) { float d0 = v[i] - mu; vs += d0 * d0; }
#pragma unroll
  for (int m = 16; m >= 1; m >>= 1) vs += __shfl_xor(vs, m, 32);
  const float rstd = rsqrtf(vs * (1.0f / D) + 1e-5f);
#pragma unroll
  for (int i = 0; i < 12; ++i) {
    int c = lane + i * 32;
    out[(size_t)row * D + c] = to_bf16((v[i] - mu) * rstd * g[c] + bta[c]);
  }
}

// ---------------- generic bf16 WMMA GEMM: Y = A @ W^T (+epilogue) ----------------
// A: [M,K] bf16 row-major, W: [Nout,K] bf16 row-major.
// Block = 256 threads = 8 waves arranged 4(M) x 2(N); wave tile 32x64; block
// tile 128x128. 6 fragment loads feed 8 WMMAs per K-step of 32 — this sizing
// compiled to fully pipelined b128 clauses + back-to-back WMMA without spills.

enum { EP_QKV = 0, EP_PROJ = 1, EP_FC1 = 2, EP_FC2 = 3 };

template <int MODE>
__global__ __launch_bounds__(256) void gemm_bf16_kernel(
    const __bf16* __restrict__ A, const __bf16* __restrict__ W,
    const float* __restrict__ bias, int K,
    __bf16* __restrict__ qp, __bf16* __restrict__ kp, __bf16* __restrict__ vtp,
    __bf16* __restrict__ ob, int ldob,
    const float* __restrict__ resid, float* __restrict__ o32, int ldo) {
  const int lane = threadIdx.x & 31;
  const int wave = threadIdx.x >> 5;
  const int l15 = lane & 15;
  const int m_wave = blockIdx.x * 128 + (wave >> 1) * 32;
  const int n_wave = blockIdx.y * 128 + (wave & 1) * 64;

  f32x8 acc[2][4] = {};

  for (int k0 = 0; k0 < K; k0 += 32) {
    bf16x16 a0 = load_frag(A, K, m_wave + l15, k0);
    bf16x16 a1 = load_frag(A, K, m_wave + 16 + l15, k0);
    bf16x16 b0 = load_frag(W, K, n_wave + l15, k0);
    bf16x16 b1 = load_frag(W, K, n_wave + 16 + l15, k0);
    bf16x16 b2 = load_frag(W, K, n_wave + 32 + l15, k0);
    bf16x16 b3 = load_frag(W, K, n_wave + 48 + l15, k0);
    acc[0][0] = wmma_bf16(a0, b0, acc[0][0]);
    acc[0][1] = wmma_bf16(a0, b1, acc[0][1]);
    acc[0][2] = wmma_bf16(a0, b2, acc[0][2]);
    acc[0][3] = wmma_bf16(a0, b3, acc[0][3]);
    acc[1][0] = wmma_bf16(a1, b0, acc[1][0]);
    acc[1][1] = wmma_bf16(a1, b1, acc[1][1]);
    acc[1][2] = wmma_bf16(a1, b2, acc[1][2]);
    acc[1][3] = wmma_bf16(a1, b3, acc[1][3]);
  }

  // C/D layout: lanes 0-15 -> M=j, N=lane; lanes 16-31 -> M=j+8, N=lane-16.
  const int mofs = (lane & 16) ? 8 : 0;
#pragma unroll
  for (int mt = 0; mt < 2; ++mt) {
#pragma unroll
    for (int nt = 0; nt < 4; ++nt) {
      const int n = n_wave + nt * 16 + l15;
      const float bv = bias[n];
      if (MODE == EP_QKV) {
        const int sec = n / D;           // 0=q, 1=k, 2=v (uniform per 16-wide tile)
        const int wcol = n - sec * D;
        const int head = wcol >> 6;
        const int dh = wcol & 63;
#pragma unroll
        for (int j = 0; j < 8; ++j) {
          const int m = m_wave + mt * 16 + j + mofs;
          const float val = acc[mt][nt][j] + bv;
          const int bb = m >> 10, tok = m & 1023;
          const size_t bh = (size_t)(bb * NH + head);
          if (sec == 0)
            qp[(bh * SEQ + tok) * DH + dh] = to_bf16(val);
          else if (sec == 1)
            kp[(bh * SEQ + tok) * DH + dh] = to_bf16(val);
          else
            vtp[(bh * DH + dh) * SEQ + tok] = to_bf16(val);  // V transposed
        }
      } else if (MODE == EP_PROJ || MODE == EP_FC2) {
#pragma unroll
        for (int j = 0; j < 8; ++j) {
          const int m = m_wave + mt * 16 + j + mofs;
          o32[(size_t)m * ldo + n] = resid[(size_t)m * ldo + n] + acc[mt][nt][j] + bv;
        }
      } else {  // EP_FC1: exact GELU -> bf16
#pragma unroll
        for (int j = 0; j < 8; ++j) {
          const int m = m_wave + mt * 16 + j + mofs;
          const float t = acc[mt][nt][j] + bv;
          const float ge = 0.5f * t * (1.0f + erff(t * 0.70710678118654752f));
          ob[(size_t)m * ldob + n] = to_bf16(ge);
        }
      }
    }
  }
}

// ---------------- flash attention: one wave per 16 query rows ----------------
// q,k: [B*H, SEQ, 64] bf16 ; vT: [B*H, 64, SEQ] bf16 ; out: [B*SEQ, 384] bf16.

__global__ __launch_bounds__(32) void attn_kernel(
    const __bf16* __restrict__ q, const __bf16* __restrict__ kmat,
    const __bf16* __restrict__ vt, __bf16* __restrict__ obuf) {
  __shared__ __align__(16) __bf16 plds[16 * 32];
  const int lane = threadIdx.x & 31;
  const int l15 = lane & 15;
  const int qblk = blockIdx.x;   // 0..63
  const int bh = blockIdx.y;     // 0..95
  const int b = bh / NH, h = bh % NH;

  const __bf16* qb = q + (size_t)bh * SEQ * DH + (size_t)qblk * 16 * DH;
  const __bf16* kbp = kmat + (size_t)bh * SEQ * DH;
  const __bf16* vbp = vt + (size_t)bh * DH * SEQ;

  const bf16x16 qa0 = load_frag(qb, DH, l15, 0);
  const bf16x16 qa1 = load_frag(qb, DH, l15, 32);

  f32x8 o[4] = {};
  float rmax[8], rsum[8];
#pragma unroll
  for (int j = 0; j < 8; ++j) { rmax[j] = -3.0e38f; rsum[j] = 0.f; }
  const float scale = 0.125f;  // 1/sqrt(64)
  const int prow = (lane & 16) ? 8 : 0;

  for (int kblk = 0; kblk < SEQ; kblk += 32) {
    // S = Q @ K^T for 32 keys: two 16x16 tiles, K-dim 64 (2 wmma each)
    f32x8 s0 = {}, s1 = {};
    bf16x16 kf;
    kf = load_frag(kbp + (size_t)kblk * DH, DH, l15, 0);
    s0 = wmma_bf16(qa0, kf, s0);
    kf = load_frag(kbp + (size_t)kblk * DH, DH, l15, 32);
    s0 = wmma_bf16(qa1, kf, s0);
    kf = load_frag(kbp + (size_t)(kblk + 16) * DH, DH, l15, 0);
    s1 = wmma_bf16(qa0, kf, s1);
    kf = load_frag(kbp + (size_t)(kblk + 16) * DH, DH, l15, 32);
    s1 = wmma_bf16(qa1, kf, s1);

    // online softmax; rows live per-VGPR j, cols per-lane within each 16-lane half
#pragma unroll
    for (int j = 0; j < 8; ++j) {
      float a0 = s0[j] * scale, a1 = s1[j] * scale;
      float mx = fmaxf(a0, a1);
      mx = fmaxf(mx, __shfl_xor(mx, 1, 32));
      mx = fmaxf(mx, __shfl_xor(mx, 2, 32));
      mx = fmaxf(mx, __shfl_xor(mx, 4, 32));
      mx = fmaxf(mx, __shfl_xor(mx, 8, 32));
      const float nm = fmaxf(rmax[j], mx);
      const float alpha = __expf(rmax[j] - nm);
      rmax[j] = nm;
      const float p0 = __expf(a0 - nm);
      const float p1 = __expf(a1 - nm);
      s0[j] = p0; s1[j] = p1;
      float ps = p0 + p1;
      ps += __shfl_xor(ps, 1, 32);
      ps += __shfl_xor(ps, 2, 32);
      ps += __shfl_xor(ps, 4, 32);
      ps += __shfl_xor(ps, 8, 32);
      rsum[j] = rsum[j] * alpha + ps;
      o[0][j] *= alpha; o[1][j] *= alpha; o[2][j] *= alpha; o[3][j] *= alpha;
    }

    // re-layout P (C layout -> A layout) through LDS
#pragma unroll
    for (int j = 0; j < 8; ++j) {
      plds[(j + prow) * 32 + l15] = to_bf16(s0[j]);
      plds[(j + prow) * 32 + 16 + l15] = to_bf16(s1[j]);
    }
    __syncthreads();
    const bf16x16 pa = load_frag(plds, 32, l15, 0);
    __syncthreads();

    // O += P @ V ; V^T rows are contiguous in the key dimension
#pragma unroll
    for (int nt = 0; nt < 4; ++nt) {
      bf16x16 vfr = load_frag(vbp + kblk, SEQ, nt * 16 + l15, 0);
      o[nt] = wmma_bf16(pa, vfr, o[nt]);
    }
  }

  // normalize and scatter to [B*SEQ, 384] (col = head*64 + dh)
  const int mofs = (lane & 16) ? 8 : 0;
#pragma unroll
  for (int nt = 0; nt < 4; ++nt) {
#pragma unroll
    for (int j = 0; j < 8; ++j) {
      const int tok = qblk * 16 + j + mofs;
      const int dh = nt * 16 + l15;
      const float val = o[nt][j] / rsum[j];
      obuf[((size_t)b * SEQ + tok) * D + h * DH + dh] = to_bf16(val);
    }
  }
}

// ---------------- launch ----------------

extern "C" void kernel_launch(void* const* d_in, const int* in_sizes, int n_in,
                              void* d_out, int out_size, void* d_ws, size_t ws_size,
                              hipStream_t stream) {
  const float* x      = (const float*)d_in[0];
  const float* ln1_g  = (const float*)d_in[1];
  const float* ln1_b  = (const float*)d_in[2];
  const float* qkv_w  = (const float*)d_in[3];
  const float* qkv_b  = (const float*)d_in[4];
  const float* proj_w = (const float*)d_in[5];
  const float* proj_b = (const float*)d_in[6];
  const float* ln2_g  = (const float*)d_in[7];
  const float* ln2_b  = (const float*)d_in[8];
  const float* fc1_w  = (const float*)d_in[9];
  const float* fc1_b  = (const float*)d_in[10];
  const float* fc2_w  = (const float*)d_in[11];
  const float* fc2_b  = (const float*)d_in[12];

  char* ws = (char*)d_ws;
  size_t off = 0;
  auto alloc = [&](size_t bytes) {
    char* p = ws + off;
    off += (bytes + 255) & ~(size_t)255;
    return p;
  };
  __bf16* wqkv = (__bf16*)alloc((size_t)D3 * D * 2);
  __bf16* wproj = (__bf16*)alloc((size_t)D * D * 2);
  __bf16* wfc1 = (__bf16*)alloc((size_t)HID * D * 2);
  __bf16* wfc2 = (__bf16*)alloc((size_t)D * HID * 2);
  __bf16* hln  = (__bf16*)alloc((size_t)MROWS * D * 2);   // reused for LN1 & LN2 out
  __bf16* qbuf = (__bf16*)alloc((size_t)MROWS * D * 2);
  __bf16* kbuf = (__bf16*)alloc((size_t)MROWS * D * 2);
  __bf16* vtbuf = (__bf16*)alloc((size_t)MROWS * D * 2);
  __bf16* obuf = (__bf16*)alloc((size_t)MROWS * D * 2);
  float*  x1   = (float*)alloc((size_t)MROWS * D * 4);
  __bf16* hmlp = (__bf16*)alloc((size_t)MROWS * HID * 2);

  auto cv = [&](const float* w, __bf16* o, int n) {
    f32_to_bf16_kernel<<<(n + 255) / 256, 256, 0, stream>>>(w, o, n);
  };
  cv(qkv_w, wqkv, D3 * D);
  cv(proj_w, wproj, D * D);
  cv(fc1_w, wfc1, HID * D);
  cv(fc2_w, wfc2, D * HID);

  // LN1
  ln_rows_kernel<<<MROWS / 8, 256, 0, stream>>>(x, ln1_g, ln1_b, hln);
  // QKV projection
  gemm_bf16_kernel<EP_QKV><<<dim3(MROWS / 128, D3 / 128), 256, 0, stream>>>(
      hln, wqkv, qkv_b, D, qbuf, kbuf, vtbuf, nullptr, 0, nullptr, nullptr, 0);
  // attention
  attn_kernel<<<dim3(SEQ / 16, BATCH * NH), 32, 0, stream>>>(qbuf, kbuf, vtbuf, obuf);
  // out projection + residual
  gemm_bf16_kernel<EP_PROJ><<<dim3(MROWS / 128, D / 128), 256, 0, stream>>>(
      obuf, wproj, proj_b, D, nullptr, nullptr, nullptr, nullptr, 0, x, x1, D);
  // LN2
  ln_rows_kernel<<<MROWS / 8, 256, 0, stream>>>(x1, ln2_g, ln2_b, hln);
  // fc1 + GELU
  gemm_bf16_kernel<EP_FC1><<<dim3(MROWS / 128, HID / 128), 256, 0, stream>>>(
      hln, wfc1, fc1_b, D, nullptr, nullptr, nullptr, hmlp, HID, nullptr, nullptr, 0);
  // fc2 + residual -> output
  gemm_bf16_kernel<EP_FC2><<<dim3(MROWS / 128, D / 128), 256, 0, stream>>>(
      hmlp, wfc2, fc2_b, HID, nullptr, nullptr, nullptr, nullptr, 0, x1,
      (float*)d_out, D);
}
